// GraphBinaryClassifier_34291018891344
// MI455X (gfx1250) — compile-verified, matrix operand order checked by reading, not run on hardware
//
#include <hip/hip_runtime.h>

typedef _Float16 v16h __attribute__((ext_vector_type(16)));
typedef float    v8f  __attribute__((ext_vector_type(8)));

// ---------------------------------------------------------------- utilities
__global__ void zero_f32(float* __restrict__ p, long long n) {
    long long i = (long long)blockIdx.x * blockDim.x + threadIdx.x;
    if (i < n) p[i] = 0.0f;
}

// ------------------------------------------------- conv1d(k=10,s=8) + ReLU
// h [N,1024] f32 -> x1h [Mpad,128] f16 (col 127 zero pad; rows >= N untouched)
__global__ void conv_relu(const float* __restrict__ h,
                          const float* __restrict__ w,
                          const float* __restrict__ b,
                          _Float16* __restrict__ x1h, int N) {
    long long idx = (long long)blockIdx.x * blockDim.x + threadIdx.x;
    int n = (int)(idx >> 7), c = (int)(idx & 127);
    if (n >= N) return;
    float v = 0.0f;
    if (c < 127) {
        const float* hp = h + (size_t)n * 1024 + c * 8;
        #pragma unroll
        for (int k = 0; k < 10; ++k) v += hp[k] * w[k];
        v += b[0];
        v = v > 0.0f ? v : 0.0f;
    }
    x1h[(size_t)n * 128 + c] = (_Float16)v;
}

// -------------------------------------- pack f32 [Kin,Nout] -> WMMA B tiles
// Per lane-element layout mirrors the ISA 16-bit A layout (M<->N):
//   lane<16 : e<8 -> K=e,      e>=8 -> K=e+8      (K 0-7, 16-23)
//   lane>=16: e<8 -> K=8+e,    e>=8 -> K=e+16     (K 8-15, 24-31)
__global__ void pack_w(const float* __restrict__ w, _Float16* __restrict__ out,
                       int Kin, int Nout) {
    int Kt = (Kin + 31) >> 5;
    int Nt = Nout >> 4;
    long long total = (long long)Kt * Nt * 512;
    long long idx = (long long)blockIdx.x * blockDim.x + threadIdx.x;
    if (idx >= total) return;
    int e    = (int)(idx & 15);
    int lane = (int)((idx >> 4) & 31);
    int tile = (int)(idx >> 9);
    int nt = tile % Nt, kt = tile / Nt;
    int col = nt * 16 + (lane & 15);
    int k   = kt * 32 + ((lane < 16) ? 0 : 8) + ((e < 8) ? e : e + 8);
    float v = (k < Kin) ? w[(size_t)k * Nout + col] : 0.0f;
    out[idx] = (_Float16)v;
}

// ---------------------------------------------------------------- degrees
__global__ void degree_kernel(const int* __restrict__ dst, float* __restrict__ deg, int E) {
    int e = blockIdx.x * blockDim.x + threadIdx.x;
    if (e < E) atomicAdd(&deg[dst[e]], 1.0f);
}

// ----------------------- edge scatter-add: agg[dst] += x[src] (f16->f32)
__global__ void scatter_add(const _Float16* __restrict__ xh,
                            const int* __restrict__ src,
                            const int* __restrict__ dst,
                            float* __restrict__ agg, int E, int F) {
    int chunks = F >> 3;
    long long t = (long long)blockIdx.x * blockDim.x + threadIdx.x;
    if (t >= (long long)E * chunks) return;
    int e = (int)(t / chunks);
    int c = (int)(t % chunks) * 8;
    int s = src[e], d = dst[e];
    const _Float16* xp = xh + (size_t)s * F + c;
    float* ap = agg + (size_t)d * F + c;
    #pragma unroll
    for (int i = 0; i < 8; ++i) atomicAdd(ap + i, (float)xp[i]);
}

// ------------------------------------- agg / max(deg,1) -> f16 activations
__global__ void norm_f16(const float* __restrict__ agg, const float* __restrict__ deg,
                         _Float16* __restrict__ out, long long total, int Fshift) {
    long long idx = (long long)blockIdx.x * blockDim.x + threadIdx.x;
    if (idx >= total) return;
    int n = (int)(idx >> Fshift);
    float dg = deg[n];
    dg = dg < 1.0f ? 1.0f : dg;
    out[idx] = (_Float16)(agg[idx] / dg);
}

// ------- fused WMMA GEMM: D = relu(A1@B1 + A2@B2 + bias), 16x64 tile / wave
// A buffers are padded to Mpad rows -> no row guard in the K loop; pad rows
// feed garbage into WMMA but only affect pad output rows, never stored.
// All four B tiles are preloaded before the WMMA burst so their loads overlap.
__global__ void gemm_fused(const _Float16* __restrict__ A1,
                           const _Float16* __restrict__ B1,
                           const _Float16* __restrict__ A2,
                           const _Float16* __restrict__ B2,
                           const float* __restrict__ bias,
                           _Float16* __restrict__ outH,
                           float* __restrict__ outF,
                           int nRows, int lda, int Kt, int Hout) {
    const int lane    = threadIdx.x;           // 0..31
    const int nt0     = blockIdx.x * 4;        // first 16-col tile of the 64-col slab
    const int rowBase = blockIdx.y * 16;       // M tile
    const int nTiles  = Hout >> 4;

    const int aRow = rowBase + (lane & 15);
    const int kOff = (lane < 16) ? 0 : 8;      // ISA 16-bit A layout

    v8f acc[4] = {};
    #pragma unroll 1
    for (int pass = 0; pass < 2; ++pass) {
        const _Float16* A = pass ? A2 : A1;
        const _Float16* B = pass ? B2 : B1;
        const _Float16* arow = A + (size_t)aRow * lda + kOff;
        #pragma unroll 1
        for (int kt = 0; kt < Kt; ++kt) {
            // A tile: two contiguous 16B chunks per lane
            v16h a;
            const _Float16* ap = arow + kt * 32;
            #pragma unroll
            for (int i = 0; i < 8; ++i) { a[i] = ap[i]; a[8 + i] = ap[16 + i]; }
            // preload all four B tiles (contiguous packed layout, 1KB apart)
            const _Float16* bp = B + (((size_t)(kt * nTiles + nt0) * 32) + lane) * 16;
            v16h bq[4];
            #pragma unroll
            for (int q = 0; q < 4; ++q)
                bq[q] = *(const v16h*)(bp + q * 512);
            // WMMA burst, A reused 4x
            #pragma unroll
            for (int q = 0; q < 4; ++q)
                acc[q] = __builtin_amdgcn_wmma_f32_16x16x32_f16(
                            false, a, false, bq[q], (short)0, acc[q], false, false);
        }
    }
    // C/D layout: VGPR j -> row rowBase + j + 8*(lane/16), col base + lane%16
    const int rAdd = (lane >> 4) * 8;
    #pragma unroll
    for (int q = 0; q < 4; ++q) {
        const int col = (nt0 + q) * 16 + (lane & 15);
        const float bv = bias[col];
        #pragma unroll
        for (int j = 0; j < 8; ++j) {
            int row = rowBase + rAdd + j;
            if (row < nRows) {
                float v = acc[q][j] + bv;
                v = v > 0.0f ? v : 0.0f;
                if (outF) outF[(size_t)row * Hout + col] = v;
                if (outH) outH[(size_t)row * Hout + col] = (_Float16)v;
            }
        }
    }
}

// ----------------------------------------------- mean pool (column sums)
__global__ void pool_kernel(const float* __restrict__ x3, float* __restrict__ hg, int N) {
    int col = threadIdx.x;                     // 128 threads
    float s = 0.0f;
    for (int r = blockIdx.x; r < N; r += gridDim.x)
        s += x3[(size_t)r * 128 + col];
    atomicAdd(&hg[col], s);
}

// -------------------------------------------------------------- MLP head
__global__ void mlp_head(const float* __restrict__ hg,
                         const float* __restrict__ w1, const float* __restrict__ b1,
                         const float* __restrict__ w2, const float* __restrict__ b2,
                         const float* __restrict__ w3, const float* __restrict__ b3,
                         float* __restrict__ out, int N) {
    __shared__ float s0[128], s1[64], s2[32];
    int t = threadIdx.x;
    if (t < 128) s0[t] = hg[t] / (float)N;
    __syncthreads();
    if (t < 64) {
        float v = b1[t];
        for (int i = 0; i < 128; ++i) v += s0[i] * w1[i * 64 + t];
        s1[t] = v > 0.0f ? v : 0.0f;
    }
    __syncthreads();
    if (t < 32) {
        float v = b2[t];
        for (int i = 0; i < 64; ++i) v += s1[i] * w2[i * 32 + t];
        s2[t] = v > 0.0f ? v : 0.0f;
    }
    __syncthreads();
    if (t == 0) {
        float v = b3[0];
        for (int i = 0; i < 32; ++i) v += s2[i] * w3[i];
        out[0] = v;
    }
}

// ----------------------------------------------------------------- driver
extern "C" void kernel_launch(void* const* d_in, const int* in_sizes, int n_in,
                              void* d_out, int out_size, void* d_ws, size_t ws_size,
                              hipStream_t stream) {
    const float* h     = (const float*)d_in[0];
    const int*   src   = (const int*)d_in[1];
    const int*   dst   = (const int*)d_in[2];
    const float* convw = (const float*)d_in[3];
    const float* convb = (const float*)d_in[4];
    const float* ws1   = (const float*)d_in[5];
    const float* wn1   = (const float*)d_in[6];
    const float* b1    = (const float*)d_in[7];
    const float* ws2   = (const float*)d_in[8];
    const float* wn2   = (const float*)d_in[9];
    const float* b2    = (const float*)d_in[10];
    const float* f1w   = (const float*)d_in[11];
    const float* f1b   = (const float*)d_in[12];
    const float* f2w   = (const float*)d_in[13];
    const float* f2b   = (const float*)d_in[14];
    const float* f3w   = (const float*)d_in[15];
    const float* f3b   = (const float*)d_in[16];

    const int N    = in_sizes[0] / 1024;
    const int E    = in_sizes[1];
    const int Mt   = (N + 15) / 16;
    const int Mpad = Mt * 16;                  // padded row count for WMMA A tiles

    // workspace carve (256B aligned)
    char* ws = (char*)d_ws;
    size_t off = 0;
    auto carve = [&](size_t bytes) -> void* {
        off = (off + 255) & ~(size_t)255;
        void* p = ws + off;
        off += bytes;
        return p;
    };
    _Float16* x1h  = (_Float16*)carve((size_t)Mpad * 128 * 2);
    _Float16* a1h  = (_Float16*)carve((size_t)Mpad * 128 * 2);
    _Float16* x2h  = (_Float16*)carve((size_t)Mpad * 256 * 2);
    _Float16* a2h  = (_Float16*)carve((size_t)Mpad * 256 * 2);
    float*    agg1 = (float*)   carve((size_t)N * 128 * 4);
    float*    agg2 = (float*)   carve((size_t)N * 256 * 4);
    float*    x3   = (float*)   carve((size_t)N * 128 * 4);
    float*    deg  = (float*)   carve((size_t)N * 4);
    float*    hg   = (float*)   carve(128 * 4);
    _Float16* w1sp = (_Float16*)carve((size_t)128 * 256 * 2);
    _Float16* w1np = (_Float16*)carve((size_t)128 * 256 * 2);
    _Float16* w2sp = (_Float16*)carve((size_t)256 * 128 * 2);
    _Float16* w2np = (_Float16*)carve((size_t)256 * 128 * 2);

    auto cdiv = [](long long a, long long b) { return (unsigned)((a + b - 1) / b); };

    // 1) zero the accumulated buffers (atomics -> must re-zero every call)
    zero_f32<<<cdiv((long long)N * 128, 256), 256, 0, stream>>>(agg1, (long long)N * 128);
    zero_f32<<<cdiv((long long)N * 256, 256), 256, 0, stream>>>(agg2, (long long)N * 256);
    zero_f32<<<cdiv(N, 256), 256, 0, stream>>>(deg, N);
    zero_f32<<<1, 128, 0, stream>>>(hg, 128);

    // 2) conv1d + relu -> f16 [N,128]
    conv_relu<<<cdiv((long long)N * 128, 256), 256, 0, stream>>>(h, convw, convb, x1h, N);

    // 3) pack weights into WMMA B layout
    pack_w<<<cdiv((long long)128 * 256, 256), 256, 0, stream>>>(ws1, w1sp, 127, 256);
    pack_w<<<cdiv((long long)128 * 256, 256), 256, 0, stream>>>(wn1, w1np, 127, 256);
    pack_w<<<cdiv((long long)256 * 128, 256), 256, 0, stream>>>(ws2, w2sp, 256, 128);
    pack_w<<<cdiv((long long)256 * 128, 256), 256, 0, stream>>>(wn2, w2np, 256, 128);

    // 4) degrees
    degree_kernel<<<cdiv(E, 256), 256, 0, stream>>>(dst, deg, E);

    // 5) SAGE layer 1: scatter, normalize, fused GEMM -> x2h [N,256] f16
    scatter_add<<<cdiv((long long)E * 16, 256), 256, 0, stream>>>(x1h, src, dst, agg1, E, 128);
    norm_f16<<<cdiv((long long)N * 128, 256), 256, 0, stream>>>(agg1, deg, a1h, (long long)N * 128, 7);
    {
        dim3 grid(256 / 64, Mt);
        gemm_fused<<<grid, 32, 0, stream>>>(x1h, w1sp, a1h, w1np, b1,
                                            x2h, (float*)nullptr, N, 128, 4, 256);
    }

    // 6) SAGE layer 2: scatter, normalize, fused GEMM -> x3 [N,128] f32
    scatter_add<<<cdiv((long long)E * 32, 256), 256, 0, stream>>>(x2h, src, dst, agg2, E, 256);
    norm_f16<<<cdiv((long long)N * 256, 256), 256, 0, stream>>>(agg2, deg, a2h, (long long)N * 256, 8);
    {
        dim3 grid(128 / 64, Mt);
        gemm_fused<<<grid, 32, 0, stream>>>(x2h, w2sp, a2h, w2np, b2,
                                            (_Float16*)nullptr, x3, N, 256, 8, 128);
    }

    // 7) mean pool + MLP head
    pool_kernel<<<256, 128, 0, stream>>>(x3, hg, N);
    mlp_head<<<1, 128, 0, stream>>>(hg, f1w, f1b, f2w, f2b, f3w, f3b, (float*)d_out, N);
}
